// HyenaOperator_34059090657949
// MI455X (gfx1250) — matrix operand order repaired and checked by reading, not compiled, not run on hardware
//
#include <hip/hip_runtime.h>
#include <hip/hip_bf16.h>

typedef __attribute__((ext_vector_type(16))) _Float16 v16h;
typedef __attribute__((ext_vector_type(8)))  _Float16 v8h;
typedef __attribute__((ext_vector_type(8)))  float    v8f;
typedef __attribute__((ext_vector_type(4)))  int      v4i;

#define DIM    768
#define LSEQ   8192
#define INNER  1536
#define NPROJ  1152
#define BATCH  2

// ---------------------------------------------------------------------------
// Optional CDNA5 async global->LDS path (ASYNCcnt), with sync fallback.
// Builtin signature (probe-confirmed by diagnostics): first param is a
// v4i pointer in the global AS, second a v4i pointer in the LDS AS.
// ---------------------------------------------------------------------------
#if defined(__has_builtin)
#if __has_builtin(__builtin_amdgcn_global_load_async_to_lds_b128) && \
    __has_builtin(__builtin_amdgcn_s_wait_asynccnt)
#define USE_ASYNC_LDS 1
#endif
#endif

#ifdef USE_ASYNC_LDS
typedef __attribute__((address_space(1))) v4i gv4i;
typedef __attribute__((address_space(3))) v4i lv4i;
__device__ __forceinline__ void async_cp16(void* lds_dst, const void* gsrc) {
  __builtin_amdgcn_global_load_async_to_lds_b128((gv4i*)gsrc, (lv4i*)lds_dst, 0, 0);
}
__device__ __forceinline__ void async_wait0() { __builtin_amdgcn_s_wait_asynccnt(0); }
#endif

// ---------------------------------------------------------------------------
// f32 -> f16 conversion (weights / long-conv filter)
// ---------------------------------------------------------------------------
__global__ __launch_bounds__(256) void cvt_f32_f16_k(const float* __restrict__ in,
                                                     _Float16* __restrict__ out, int n) {
  int i = blockIdx.x * 256 + threadIdx.x;
  if (i < n) out[i] = (_Float16)in[i];
}

// ---------------------------------------------------------------------------
// WMMA GEMM: C[M x N] = A[M x K] * B^T + bias   (B row-major [N x K])
// 256 threads = 8 waves; each wave owns a 16x16 tile; block = 128 rows x 16
// cols. The 16xK B strip (contiguous rows) is staged once in LDS; fragments
// are 32B-aligned v16h LDS loads. A is streamed from global with prefetch.
// A-fragment K index (contiguous k order):  K = j + 16*hi
// B-fragment K index (contiguous, same as A):  K = j + 16*hi
// ---------------------------------------------------------------------------
template <typename TA, int K>
__global__ __launch_bounds__(256) void gemm_wmma_k(const TA* __restrict__ A,
                                                   const _Float16* __restrict__ Bh,
                                                   const float* __restrict__ bias,
                                                   float* __restrict__ C, int N) {
  __shared__ __align__(32) _Float16 sB[16 * K];
  const int lane = threadIdx.x & 31;
  const int wave = threadIdx.x >> 5;
  const int hi   = lane >> 4;
  const int lo   = lane & 15;
  const int m0   = blockIdx.x * 128 + wave * 16;
  const int n0   = blockIdx.y * 16;

  {  // stage B strip: 16*K contiguous halfs at Bh + n0*K
    const uint4* gB = (const uint4*)(Bh + (size_t)n0 * K);
    uint4* lB = (uint4*)sB;
    const int n16 = (16 * K) / 8;
#ifdef USE_ASYNC_LDS
    for (int i = threadIdx.x; i < n16; i += 256) async_cp16(lB + i, gB + i);
    async_wait0();
#else
    for (int i = threadIdx.x; i < n16; i += 256) lB[i] = gB[i];
#endif
  }
  __syncthreads();

  const TA*       arow = A  + (size_t)(m0 + lo) * K;
  const _Float16* brow = sB + lo * K;

  v8f acc = {};
  for (int k0 = 0; k0 < K; k0 += 32) {
    __builtin_prefetch(arow + k0 + 64, 0, 1);   // global_prefetch on A stream
    v16h a;
#pragma unroll
    for (int j = 0; j < 8; ++j) {
      a[j]     = (_Float16)arow[k0 + 16 * hi + j];
      a[j + 8] = (_Float16)arow[k0 + 16 * hi + 8 + j];
    }
    const v16h b = *reinterpret_cast<const v16h*>(brow + k0 + 16 * hi);
    acc = __builtin_amdgcn_wmma_f32_16x16x32_f16(false, a, false, b,
                                                 (short)0, acc, false, false);
  }
  const float bv = bias[n0 + lo];
#pragma unroll
  for (int e = 0; e < 8; ++e) {
    C[(size_t)(m0 + e + 8 * hi) * N + (n0 + lo)] = acc[e] + bv;
  }
}

// ---------------------------------------------------------------------------
// Depthwise causal short conv (K=3, left pad 2) + tied-group repeat + pre-gate.
// up: [B, L, NPROJ] f32; channel ch of TOTAL -> up column ch % NPROJ.
// Each thread produces 8 consecutive t for one channel (rolling registers),
// writing one 16B v8h per output tensor. Outputs channel-major f16 [B,INNER,L].
// ---------------------------------------------------------------------------
__global__ __launch_bounds__(256) void shortconv_gate_k(const float* __restrict__ up,
                                                        const float* __restrict__ sw,
                                                        const float* __restrict__ sb,
                                                        _Float16* __restrict__ vpre,
                                                        _Float16* __restrict__ x0h) {
  const size_t idx = (size_t)blockIdx.x * 256 + threadIdx.x;  // c fastest
  const int    c  = (int)(idx % INNER);
  const size_t r  = idx / INNER;
  const int    t0 = (int)(r % (LSEQ / 8)) * 8;
  const int    b  = (int)(r / (LSEQ / 8));

  int   col[3];
  float w0[3], w1[3], w2[3], bb[3], um2[3], um1[3];
#pragma unroll
  for (int q = 0; q < 3; ++q) {
    const int ch = c + q * INNER;
    col[q] = ch % NPROJ;
    w0[q] = sw[ch * 3 + 0];
    w1[q] = sw[ch * 3 + 1];
    w2[q] = sw[ch * 3 + 2];
    bb[q] = sb[ch];
    um2[q] = (t0 >= 2) ? up[((size_t)b * LSEQ + t0 - 2) * NPROJ + col[q]] : 0.0f;
    um1[q] = (t0 >= 1) ? up[((size_t)b * LSEQ + t0 - 1) * NPROJ + col[q]] : 0.0f;
  }

  v8h vv, xx;
#pragma unroll
  for (int i = 0; i < 8; ++i) {
    const int t = t0 + i;
    float uc[3];
#pragma unroll
    for (int q = 0; q < 3; ++q) {
      const float cur = up[((size_t)b * LSEQ + t) * NPROJ + col[q]];
      uc[q] = bb[q] + w0[q] * um2[q] + w1[q] * um1[q] + w2[q] * cur;
      um2[q] = um1[q];
      um1[q] = cur;
    }
    vv[i] = (_Float16)(uc[2] * uc[1]);   // pre-gate v * x1
    xx[i] = (_Float16)uc[0];
  }
  const size_t o = ((size_t)b * INNER + c) * LSEQ + t0;
  *reinterpret_cast<v8h*>(vpre + o) = vv;
  *reinterpret_cast<v8h*>(x0h  + o) = xx;
}

// ---------------------------------------------------------------------------
// Causal long conv as block-Toeplitz WMMA + fused bias-skip and post-gate.
// Block = one channel c, FOUR 256-step output groups g = 4G..4G+3 (WMMA N dim
// batches 16 output blocks per group). For a delay-block pair (d, d+1):
//   A[i][K]: Toeplitz from k (shared by all 4 groups -> built once per d)
//   B[K][n]: shifted v; contiguous 32B-aligned v16h LDS load on the fast
//            path, guarded scalar path only at causal boundaries.
// 8 waves split delay pairs, 4 accumulators each; f32 cross-wave reduction in
// LDS; epilogue adds conv_bias*v, multiplies x0, writes time-major f16 y.
// ---------------------------------------------------------------------------
__global__ __launch_bounds__(256) void longconv_wmma_k(const _Float16* __restrict__ kh,
                                                       const _Float16* __restrict__ vpre,
                                                       const _Float16* __restrict__ x0h,
                                                       const float* __restrict__ conv_bias,
                                                       _Float16* __restrict__ yh) {
  __shared__ __align__(32) _Float16 sk[LSEQ];   // 16 KB
  __shared__ __align__(32) _Float16 sv[LSEQ];   // 16 KB
  __shared__ float sred[8192];                  // 32 KB: 4 groups x 8 waves x 256

  const int G    = blockIdx.x;                  // 0..7 -> groups 4G..4G+3
  const int bc   = blockIdx.y;
  const int b    = bc / INNER;
  const int c    = bc % INNER;
  const int lane = threadIdx.x & 31;
  const int wave = threadIdx.x >> 5;
  const int hi   = lane >> 4;
  const int lo   = lane & 15;
  const int gtop = 4 * G + 3;
  const int nkv  = 256 * (gtop + 1);            // staged halfs, <= 8192

  {  // stage k row and v prefix
    const uint4* gk = (const uint4*)(kh   + (size_t)c * LSEQ);
    const uint4* gv = (const uint4*)(vpre + ((size_t)b * INNER + c) * LSEQ);
    uint4* lk = (uint4*)sk;
    uint4* lv = (uint4*)sv;
    const int n16 = nkv >> 3;
#ifdef USE_ASYNC_LDS
    for (int i = threadIdx.x; i < n16; i += 256) {
      async_cp16(lk + i, gk + i);
      async_cp16(lv + i, gv + i);
    }
    async_wait0();
#else
    for (int i = threadIdx.x; i < n16; i += 256) { lk[i] = gk[i]; lv[i] = gv[i]; }
#endif
  }
  __syncthreads();

  v8f acc0 = {}, acc1 = {}, acc2 = {}, acc3 = {};
  const int pend = 8 * (gtop + 1);
  for (int p = wave; p < pend; p += 8) {
    const int dlt = 2 * p;                      // delta pair (dlt, dlt+1) => K=32
    // --- A: Toeplitz fragment from k, built once, used by 4 WMMAs ---
    // run j=0..7:  d = 16*dlt + lo + 16*hi - j
    // run j=8..15: d = 16*dlt + lo + 16*hi - j
    v16h a;
    if (dlt >= 1) {
#pragma unroll
      for (int j = 0; j < 8; ++j) {
        a[j]     = sk[16 * dlt + lo + 16 * hi - j];
        a[j + 8] = sk[16 * dlt + lo + 16 * hi - 8 - j];
      }
    } else {
#pragma unroll
      for (int j = 0; j < 8; ++j) {
        const int d = lo + 16 * hi - j;
        a[j]     = (d >= 0) ? sk[d] : (_Float16)0.0f;
        a[j + 8] = (d >= 8) ? sk[d - 8] : (_Float16)0.0f;
      }
    }
    // --- B per group: contiguous aligned run of 16 halfs of v ---
#pragma unroll
    for (int gx = 0; gx < 4; ++gx) {
      const int g    = 4 * G + gx;
      const int base = 256 * g - 16 * dlt + 16 * lo - 16 * hi;
      v16h bf;
      if (16 * dlt <= 256 * g - 16) {           // uniform fast path: run fully >= 0
        bf = *reinterpret_cast<const v16h*>(sv + base);
      } else {
#pragma unroll
        for (int j = 0; j < 16; ++j) {
          const int s = base + j;
          bf[j] = (s >= 0) ? sv[s] : (_Float16)0.0f;
        }
      }
      v8f* accp = (gx == 0) ? &acc0 : (gx == 1) ? &acc1 : (gx == 2) ? &acc2 : &acc3;
      *accp = __builtin_amdgcn_wmma_f32_16x16x32_f16(false, a, false, bf,
                                                     (short)0, *accp, false, false);
    }
  }

  {  // cross-wave reduction staging
    float* my = sred + wave * 256 + lane * 8;
#pragma unroll
    for (int e = 0; e < 8; ++e) {
      my[e]        = acc0[e];
      my[e + 2048] = acc1[e];
      my[e + 4096] = acc2[e];
      my[e + 6144] = acc3[e];
    }
  }
  __syncthreads();
  {
    const int tid = threadIdx.x;                // one thread per (M,N) of 16x16 tile
    const int M = tid >> 4;
    const int N = tid & 15;
    const int srcLane = N + ((M >= 8) ? 16 : 0);
    const int e = M & 7;
    const float cb = conv_bias[c];
#pragma unroll
    for (int gx = 0; gx < 4; ++gx) {
      float sum = 0.0f;
#pragma unroll
      for (int w = 0; w < 8; ++w) sum += sred[gx * 2048 + w * 256 + srcLane * 8 + e];
      const int t = 256 * (4 * G + gx) + 16 * N + M;
      const float vv  = (float)sv[t];
      const float x0v = (float)x0h[((size_t)b * INNER + c) * LSEQ + t];
      const float yv  = (sum + cb * vv) * x0v;
      yh[((size_t)b * LSEQ + t) * INNER + c] = (_Float16)yv;
    }
  }
}

// ---------------------------------------------------------------------------
// Launch
// ---------------------------------------------------------------------------
extern "C" void kernel_launch(void* const* d_in, const int* in_sizes, int n_in,
                              void* d_out, int out_size, void* d_ws, size_t ws_size,
                              hipStream_t stream) {
  const float* u       = (const float*)d_in[0];   // [2, 8192, 768]
  const float* w_in    = (const float*)d_in[1];   // [1152, 768]
  const float* b_in    = (const float*)d_in[2];   // [1152]
  const float* short_w = (const float*)d_in[3];   // [4608, 1, 3]
  const float* short_b = (const float*)d_in[4];   // [4608]
  const float* kfil    = (const float*)d_in[5];   // [1536, 8192]
  const float* conv_b  = (const float*)d_in[6];   // [1536]
  const float* w_out   = (const float*)d_in[7];   // [768, 1536]
  const float* b_out   = (const float*)d_in[8];   // [768]
  float* out = (float*)d_out;                     // [2, 8192, 768] f32

  char* p = (char*)d_ws;
  auto take = [&](size_t bytes) {
    char* r = p;
    p += (bytes + 255) & ~(size_t)255;
    return r;
  };
  float*    ws_up   = (float*)   take(sizeof(float) * (size_t)BATCH * LSEQ * NPROJ);
  _Float16* ws_vpre = (_Float16*)take(2ull * BATCH * INNER * LSEQ);
  _Float16* ws_x0   = (_Float16*)take(2ull * BATCH * INNER * LSEQ);
  _Float16* ws_y    = (_Float16*)take(2ull * BATCH * LSEQ * INNER);
  _Float16* w1h     = (_Float16*)take(2ull * NPROJ * DIM);
  _Float16* w2h     = (_Float16*)take(2ull * DIM * INNER);
  _Float16* kh      = (_Float16*)take(2ull * INNER * LSEQ);

  {  // weight / filter narrowing to f16
    int n1 = NPROJ * DIM;
    cvt_f32_f16_k<<<(n1 + 255) / 256, 256, 0, stream>>>(w_in, w1h, n1);
    int n2 = DIM * INNER;
    cvt_f32_f16_k<<<(n2 + 255) / 256, 256, 0, stream>>>(w_out, w2h, n2);
    int n3 = INNER * LSEQ;
    cvt_f32_f16_k<<<(n3 + 255) / 256, 256, 0, stream>>>(kfil, kh, n3);
  }

  // in_proj GEMM: [16384 x 768] * [768 x 1152] + bias -> ws_up (f32)
  gemm_wmma_k<float, DIM><<<dim3((BATCH * LSEQ) / 128, NPROJ / 16), 256, 0, stream>>>(
      u, w1h, b_in, ws_up, NPROJ);

  // short conv + gating -> ws_vpre, ws_x0 (channel-major f16)
  {
    size_t total = (size_t)BATCH * INNER * (LSEQ / 8);
    shortconv_gate_k<<<(unsigned)(total / 256), 256, 0, stream>>>(
        ws_up, short_w, short_b, ws_vpre, ws_x0);
  }

  // block-Toeplitz long conv + bias skip + post-gate -> ws_y [B, L, INNER] f16
  longconv_wmma_k<<<dim3(LSEQ / 1024, BATCH * INNER), 256, 0, stream>>>(
      kh, ws_vpre, ws_x0, conv_b, ws_y);

  // out GEMM: [16384 x 1536] * [1536 x 768] + bias -> d_out (f32)
  gemm_wmma_k<_Float16, INNER><<<dim3((BATCH * LSEQ) / 128, DIM / 16), 256, 0, stream>>>(
      ws_y, w2h, b_out, out, DIM);
}
